// CrossAttention_62027917689453
// MI455X (gfx1250) — compile-verified, matrix operand order checked by reading, not run on hardware
//
#include <hip/hip_runtime.h>
#include <hip/hip_bf16.h>

typedef __bf16 bf16_t;
typedef __attribute__((ext_vector_type(16))) __bf16 v16bf;
typedef __attribute__((ext_vector_type(8)))  float  v8f;
typedef __attribute__((ext_vector_type(4)))  unsigned int u32x4;
typedef __attribute__((ext_vector_type(8)))  int i32x8;
typedef __attribute__((ext_vector_type(4)))  int i32x4;

#ifndef __has_builtin
#define __has_builtin(x) 0
#endif
#if __has_builtin(__builtin_amdgcn_tensor_load_to_lds) && \
    __has_builtin(__builtin_amdgcn_s_wait_tensorcnt)
#define HAVE_TDM 1
#else
#define HAVE_TDM 0
#endif

#define WMMA_BF16(a, b, c)                                                     \
  __builtin_amdgcn_wmma_f32_16x16x32_bf16(false, (a), false, (b), (short)0,    \
                                          (c), false, false)

// ---------------------------------------------------------------------------
// Fragment loader (wave32), A and B fragments (B tiles stored N-major).
// elements 0..7  : K = half*8 + 0..7      (16B run)
// elements 8..15 : K = 16 + half*8 + 0..7 (16B run)
// ---------------------------------------------------------------------------
__device__ __forceinline__ v16bf load_frag_rm(const bf16_t* __restrict__ p,
                                              int row0, int stride, int k0) {
  const int lane = threadIdx.x & 31;
  const int m = lane & 15, half = lane >> 4;
  const bf16_t* q = p + (size_t)(row0 + m) * stride + k0 + half * 8;
  union { uint4 u; bf16_t h[8]; } lo, hi;
  lo.u = *reinterpret_cast<const uint4*>(q);
  hi.u = *reinterpret_cast<const uint4*>(q + 16);
  v16bf f;
#pragma unroll
  for (int e = 0; e < 8; ++e) { f[e] = lo.h[e]; f[e + 8] = hi.h[e]; }
  return f;
}

// ---------------------------------------------------------------------------
// Generic tiled transpose: dst[C][R](bf16) = src[R][C](f32). 64x64 tiles.
// Used once per weight matrix and once per batch of freq (V^T).
// ---------------------------------------------------------------------------
__global__ void __launch_bounds__(256)
transpose_f32_to_bf16(const float* __restrict__ src, bf16_t* __restrict__ dst,
                      int R, int C) {
  __shared__ __align__(16) bf16_t tile[64][64 + 8];
  const int tiles_c = C >> 6;
  const int rt = blockIdx.x / tiles_c, ct = blockIdx.x % tiles_c;
  const int r0 = rt * 64, c0 = ct * 64;
  const int tid = threadIdx.x;

#pragma unroll
  for (int i = 0; i < 4; ++i) {
    int idx = i * 256 + tid;                   // 1024 float4 groups
    int r = idx >> 4, cg = (idx & 15) * 4;
    float4 f = *reinterpret_cast<const float4*>(
        &src[(size_t)(r0 + r) * C + c0 + cg]);
    tile[r][cg + 0] = (bf16_t)f.x;
    tile[r][cg + 1] = (bf16_t)f.y;
    tile[r][cg + 2] = (bf16_t)f.z;
    tile[r][cg + 3] = (bf16_t)f.w;
  }
  __syncthreads();
#pragma unroll
  for (int i = 0; i < 2; ++i) {
    int idx = i * 256 + tid;                   // 512 groups of 8 bf16
    int cr = idx >> 3, rg = (idx & 7) * 8;
    union { uint4 u; bf16_t h[8]; } pk;
#pragma unroll
    for (int e = 0; e < 8; ++e) pk.h[e] = tile[rg + e][cr];
    *reinterpret_cast<uint4*>(&dst[(size_t)(c0 + cr) * R + r0 + rg]) = pk.u;
  }
}

// ---------------------------------------------------------------------------
// Kernel 1: C[M,Nc](bf16) = A[M,K](f32) * W[K,Nc]. W passed pre-transposed
// as Wt[Nc][K] bf16 in global (L2-resident), so B-frags are global b128 pairs.
// 128x128 tiles, 8 waves (4x2), K-steps of 32; only A staged in LDS.
// ---------------------------------------------------------------------------
#define PM 128
#define PN 128
#define PK 32
#define PKP (PK + 8)

__global__ void __launch_bounds__(256)
proj_gemm_bf16(const float* __restrict__ A, const bf16_t* __restrict__ Wt,
               bf16_t* __restrict__ C, int M, int K, int Nc) {
  __shared__ __align__(16) bf16_t As[PM][PKP];

  const int tiles_n = Nc / PN;
  const int bm = blockIdx.x / tiles_n, bn = blockIdx.x % tiles_n;
  const int row0 = bm * PM, col0 = bn * PN;
  const int tid = threadIdx.x;
  const int wave = tid >> 5, lane = tid & 31;
  const int wr = wave >> 1, wc = wave & 1;
  const int nn = lane & 15, half = lane >> 4;

  v8f acc[2][4];
#pragma unroll
  for (int i = 0; i < 2; ++i)
#pragma unroll
    for (int j = 0; j < 4; ++j) acc[i][j] = {};

  for (int kk = 0; kk < K; kk += PK) {
    // stage A tile 128x32: float4 reads, b128 LDS stores (8 bf16 each)
#pragma unroll
    for (int i = 0; i < 2; ++i) {
      int idx = i * 256 + tid;                 // 512 groups of 8
      int r = idx >> 2, cg = (idx & 3) * 8;
      const float* ap = &A[(size_t)(row0 + r) * K + kk + cg];
      if (kk + PK < K) __builtin_prefetch(ap + PK, 0, 3);
      float4 f0 = *reinterpret_cast<const float4*>(ap);
      float4 f1 = *reinterpret_cast<const float4*>(ap + 4);
      union { uint4 u; bf16_t h[8]; } pk;
      pk.h[0] = (bf16_t)f0.x; pk.h[1] = (bf16_t)f0.y;
      pk.h[2] = (bf16_t)f0.z; pk.h[3] = (bf16_t)f0.w;
      pk.h[4] = (bf16_t)f1.x; pk.h[5] = (bf16_t)f1.y;
      pk.h[6] = (bf16_t)f1.z; pk.h[7] = (bf16_t)f1.w;
      *reinterpret_cast<uint4*>(&As[r][cg]) = pk.u;
    }
    __syncthreads();

    v16bf af[2];
#pragma unroll
    for (int i = 0; i < 2; ++i)
      af[i] = load_frag_rm(&As[0][0], wr * 32 + i * 16, PKP, 0);
#pragma unroll
    for (int j = 0; j < 4; ++j) {
      v16bf bf = load_frag_rm(Wt, col0 + wc * 64 + j * 16, K, kk);
#pragma unroll
      for (int i = 0; i < 2; ++i) acc[i][j] = WMMA_BF16(af[i], bf, acc[i][j]);
    }
    __syncthreads();
  }

#pragma unroll
  for (int i = 0; i < 2; ++i)
#pragma unroll
    for (int j = 0; j < 4; ++j) {
      int rbase = row0 + wr * 32 + i * 16 + half * 8;
      int cbase = col0 + wc * 64 + j * 16 + nn;
#pragma unroll
      for (int v = 0; v < 8; ++v)
        C[(size_t)(rbase + v) * Nc + cbase] = (bf16_t)acc[i][j][v];
    }
}

// ---------------------------------------------------------------------------
// Kernel 2: out[:, :, 0:1024] = rgb (float4 copy)
// ---------------------------------------------------------------------------
__global__ void __launch_bounds__(256)
copy_rgb_kernel(const float* __restrict__ rgb, float* __restrict__ out) {
  const size_t total4 = (size_t)8 * 2048 * 1024 / 4;
  for (size_t i = (size_t)blockIdx.x * blockDim.x + threadIdx.x; i < total4;
       i += (size_t)gridDim.x * blockDim.x) {
    size_t flat = i * 4;
    size_t row = flat >> 10, dcol = flat & 1023;
    *reinterpret_cast<float4*>(out + row * 2048 + dcol) =
        reinterpret_cast<const float4*>(rgb)[i];
  }
}

// ---------------------------------------------------------------------------
// Kernel 3: flash attention. grid = B * (N/16) * 2.
// Q tile (16x1024 bf16) staged to LDS by the Tensor Data Mover; TDM padding
// (4 dwords per 256 dwords) gives a bank-staggered layout: row stride 1040.
// ---------------------------------------------------------------------------
#define BM 16
#define BN 128
#define DHALF 512
#define QROW 1040
#define QCHUNK 520
#define PSP (BN + 8)

__device__ __forceinline__ v16bf load_frag_qpad(const bf16_t* __restrict__ qs,
                                                int k0) {
  const int lane = threadIdx.x & 31;
  const int m = lane & 15, half = lane >> 4;
  const int off = m * QROW + (k0 >> 9) * QCHUNK + (k0 & 511) + half * 8;
  union { uint4 u; bf16_t h[8]; } lo, hi;
  lo.u = *reinterpret_cast<const uint4*>(qs + off);
  hi.u = *reinterpret_cast<const uint4*>(qs + off + 16);
  v16bf f;
#pragma unroll
  for (int e = 0; e < 8; ++e) { f[e] = lo.h[e]; f[e + 8] = hi.h[e]; }
  return f;
}

__global__ void __launch_bounds__(256)
flash_attn(const bf16_t* __restrict__ Qg, const bf16_t* __restrict__ Kg,
           const bf16_t* __restrict__ Vt, float* __restrict__ out) {
  __shared__ __align__(16) bf16_t Qs[BM * QROW];  // 32.5 KB, TDM-padded
  __shared__ __align__(16) bf16_t Ps[BM][PSP];
  __shared__ float m_run[BM], l_run[BM], corr_s[BM], mnew_s[BM];
  __shared__ float wred[8][BM];

  const int blk = blockIdx.x;
  const int dh = blk & 1;
  const int qb = (blk >> 1) & 127;
  const int b  = blk >> 8;
  const int qrow0 = qb * BM;
  const size_t Nn = 2048, Dd = 1024;
  const bf16_t* Qb  = Qg + (size_t)b * Nn * Dd;
  const bf16_t* Kb  = Kg + (size_t)b * Nn * Dd;
  const bf16_t* Vtb = Vt + (size_t)b * Dd * Nn;  // [d][m]

  const int tid = threadIdx.x, lane = tid & 31, wave = tid >> 5;
  const int nn = lane & 15, half = lane >> 4;
  const float scale = 0.03125f;

  if (tid < BM) { m_run[tid] = -1e30f; l_run[tid] = 0.0f; }

#if HAVE_TDM
  if (wave == 0) {   // wave-uniform -> scalar branch; TDM ignores EXEC
    unsigned long long ga =
        (unsigned long long)(size_t)(&Qb[(size_t)qrow0 * Dd]);
    unsigned lds_base = (unsigned)(size_t)(void*)&Qs[0];
    u32x4 g0;
    g0[0] = 1u;                                    // count=1
    g0[1] = lds_base;                              // lds_addr
    g0[2] = (unsigned)(ga & 0xFFFFFFFFull);        // global_addr[31:0]
    g0[3] = (unsigned)(ga >> 32) | (2u << 30);     // addr[56:32] | type=2
    i32x8 g1;
    g1[0] = (1 << 16) | (1 << 20) | (7 << 22) | (3 << 25);
    g1[1] = (int)(1024u << 16);                    // tensor_dim0 = 1024
    g1[2] = (int)(2048u << 16);                    // tensor_dim1 = 2048
    g1[3] = (int)(1024u << 16);                    // tile_dim0 = 1024
    g1[4] = 16;                                    // tile_dim1 = 16
    g1[5] = 1024;                                  // tensor_dim0_stride
    g1[6] = 0;
    g1[7] = 0;
    i32x4 gz = {0, 0, 0, 0};
#if __clang_major__ >= 23
    i32x8 gz8 = {0, 0, 0, 0, 0, 0, 0, 0};
    __builtin_amdgcn_tensor_load_to_lds(g0, g1, gz, gz, gz8, 0);
#else
    __builtin_amdgcn_tensor_load_to_lds(g0, g1, gz, gz, 0);
#endif
    __builtin_amdgcn_s_wait_tensorcnt(0);
  }
#else
#pragma unroll
  for (int i = 0; i < 8; ++i) {
    int idx = i * 256 + tid;
    int r = idx >> 7, c = (idx & 127) * 8;
    int off = r * QROW + (c >> 9) * QCHUNK + (c & 511);
    *reinterpret_cast<uint4*>(&Qs[off]) =
        *reinterpret_cast<const uint4*>(&Qb[(size_t)(qrow0 + r) * Dd + c]);
  }
#endif
  __syncthreads();

  v8f o_acc[4];
#pragma unroll
  for (int t = 0; t < 4; ++t) o_acc[t] = {};

  for (int kb = 0; kb < (int)Nn; kb += BN) {
    if (kb + BN < (int)Nn) {
      __builtin_prefetch(&Kb[(size_t)(kb + BN + wave * 16 + nn) * Dd], 0, 3);
      __builtin_prefetch(
          &Vtb[(size_t)(dh * DHALF + wave * 64 + nn) * Nn + kb + BN], 0, 3);
    }

    // ---- S subtile
    v8f s = {};
    for (int d = 0; d < (int)Dd; d += 32) {
      v16bf aq = load_frag_qpad(&Qs[0], d);
      v16bf bk = load_frag_rm(Kb, kb + wave * 16, Dd, d);
      s = WMMA_BF16(aq, bk, s);
    }

    // ---- row max within this wave's 16 key-cols
#pragma unroll
    for (int v = 0; v < 8; ++v) {
      float r = s[v] * scale;
      s[v] = r;
#pragma unroll
      for (int off = 1; off < 16; off <<= 1)
        r = fmaxf(r, __shfl_xor(r, off, 32));
      if (nn == 0) wred[wave][half * 8 + v] = r;
    }
    __syncthreads();
    if (tid < BM) {
      float bm_ = wred[0][tid];
#pragma unroll
      for (int w2 = 1; w2 < 8; ++w2) bm_ = fmaxf(bm_, wred[w2][tid]);
      float mo = m_run[tid];
      float mn = fmaxf(mo, bm_);
      mnew_s[tid] = mn;
      corr_s[tid] = __expf(mo - mn);
      m_run[tid] = mn;
    }
    __syncthreads();

    // ---- P = exp(S - m_new); stage; row sums; rescale O
#pragma unroll
    for (int v = 0; v < 8; ++v) {
      int row = half * 8 + v;
      float p = __expf(s[v] - mnew_s[row]);
      Ps[row][wave * 16 + nn] = (bf16_t)p;
      float rs = p;
#pragma unroll
      for (int off = 1; off < 16; off <<= 1) rs += __shfl_xor(rs, off, 32);
      if (nn == 0) wred[wave][row] = rs;
    }
#pragma unroll
    for (int t = 0; t < 4; ++t)
#pragma unroll
      for (int v = 0; v < 8; ++v) o_acc[t][v] *= corr_s[half * 8 + v];
    __syncthreads();
    if (tid < BM) {
      float ssum = 0.0f;
#pragma unroll
      for (int w2 = 0; w2 < 8; ++w2) ssum += wred[w2][tid];
      l_run[tid] = l_run[tid] * corr_s[tid] + ssum;
    }
    __syncthreads();

    // ---- O += P . V ; B-frags straight from global V^T
#pragma unroll
    for (int kc = 0; kc < BN; kc += 32) {
      v16bf ap = load_frag_rm(&Ps[0][0], 0, PSP, kc);
#pragma unroll
      for (int t = 0; t < 4; ++t) {
        v16bf bv = load_frag_rm(Vtb, dh * DHALF + wave * 64 + t * 16,
                                (int)Nn, kb + kc);
        o_acc[t] = WMMA_BF16(ap, bv, o_acc[t]);
      }
    }
    __syncthreads();
  }

  // ---- epilogue
#pragma unroll
  for (int t = 0; t < 4; ++t)
#pragma unroll
    for (int v = 0; v < 8; ++v) {
      int row = half * 8 + v;
      float val = o_acc[t][v] * (0.5f / l_run[row]);
      size_t orow = (size_t)b * 2048 + qrow0 + row;
      out[orow * 2048 + 1024 + dh * DHALF + wave * 64 + t * 16 + nn] = val;
    }
}

// ---------------------------------------------------------------------------
extern "C" void kernel_launch(void* const* d_in, const int* in_sizes, int n_in,
                              void* d_out, int out_size, void* d_ws,
                              size_t ws_size, hipStream_t stream) {
  const float* rgb  = (const float*)d_in[0];
  const float* freq = (const float*)d_in[1];
  // d_in[2] = ifreq, d_in[5] = Wv: dead code in reference
  const float* Wq = (const float*)d_in[3];
  const float* Wk = (const float*)d_in[4];
  float* out = (float*)d_out;

  const size_t MBN = (size_t)8 * 2048;   // 16384
  const size_t D = 1024;
  bf16_t* Qw  = (bf16_t*)d_ws;           // 32 MB
  bf16_t* Kw  = Qw + MBN * D;            // 32 MB
  bf16_t* VtW = Kw + MBN * D;            // 32 MB, [b][d][m]
  bf16_t* WqT = VtW + MBN * D;           // 2 MB, [e][d]
  bf16_t* WkT = WqT + D * D;             // 2 MB, [e][d]

  // one-time weight transposes (f32 -> bf16, N-major)
  transpose_f32_to_bf16<<<dim3(16 * 16), 256, 0, stream>>>(Wq, WqT, 1024, 1024);
  transpose_f32_to_bf16<<<dim3(16 * 16), 256, 0, stream>>>(Wk, WkT, 1024, 1024);
  // per-batch freq transposes -> V^T
  for (int b = 0; b < 8; ++b)
    transpose_f32_to_bf16<<<dim3(32 * 16), 256, 0, stream>>>(
        freq + (size_t)b * 2048 * 1024, VtW + (size_t)b * 1024 * 2048,
        2048, 1024);

  proj_gemm_bf16<<<dim3((MBN / PM) * (D / PN)), 256, 0, stream>>>(
      rgb, WqT, Qw, (int)MBN, (int)D, (int)D);
  proj_gemm_bf16<<<dim3((MBN / PM) * (D / PN)), 256, 0, stream>>>(
      freq, WkT, Kw, (int)MBN, (int)D, (int)D);

  copy_rgb_kernel<<<dim3(2048), 256, 0, stream>>>(rgb, out);

  flash_attn<<<dim3(8 * 128 * 2), 256, 0, stream>>>(Qw, Kw, VtW, out);
}
